// LinAttention_8014408974650
// MI455X (gfx1250) — matrix operand order repaired and checked by reading, not compiled, not run on hardware
//
#include <hip/hip_runtime.h>
#include <hip/hip_bf16.h>
#include <stdint.h>

#define D_DIM  128
#define KQ_DIM 64
#define H_NUM  8
#define B_NUM  32
#define N_LEN  2048
#define NSPLIT 4

typedef __attribute__((ext_vector_type(16))) __bf16       v16bf;
typedef __attribute__((ext_vector_type(8)))  float        v8f;
typedef __attribute__((ext_vector_type(4)))  float        v4f;
typedef __attribute__((ext_vector_type(4)))  unsigned int v4u;
typedef __attribute__((ext_vector_type(2)))  unsigned int v2u;

union FragU  { v4u q[2]; v16bf v; };
union Pack4  { __bf16 h[4]; v2u u; };
union Pack8  { __bf16 h[8]; v4u u; };

// ---- WMMA fragment loaders (bf16 16x16x32), per CDNA5 ISA VGPR layouts ----
// A (16x32, MxK): lanes 0-15 -> M=lane, K in {k0..k0+7} U {k0+16..k0+23};
//                 lanes 16-31 -> K in {k0+8..k0+15} U {k0+24..k0+31}.
// Requires LDS layout [row=M][K contiguous].
__device__ __forceinline__ v16bf load_frag_a(const __bf16* base, int stride, int row0, int k0) {
  const int lane = threadIdx.x & 31;
  const int hi = lane >> 4;
  const int r = lane & 15;
  const __bf16* p = base + (row0 + r) * stride + k0 + hi * 8;
  FragU f;
  f.q[0] = *(const v4u*)(p);        // 8 bf16 (16B)
  f.q[1] = *(const v4u*)(p + 16);   // 8 bf16 at K+16
  return f.v;
}

// B (32x16, KxN): lanes 0-15 -> N=lane, K in {k0..k0+7} U {k0+16..k0+23};
//                 lanes 16-31 -> K in {k0+8..k0+15} U {k0+24..k0+31}.
// Requires LDS layout [row=N][K contiguous]  (i.e. B^T staged).
__device__ __forceinline__ v16bf load_frag_b(const __bf16* base, int stride, int col0, int k0) {
  const int lane = threadIdx.x & 31;
  const int hi = lane >> 4;
  const int c = lane & 15;
  const __bf16* p = base + (col0 + c) * stride + k0 + hi * 8;
  FragU f;
  f.q[0] = *(const v4u*)(p);
  f.q[1] = *(const v4u*)(p + 16);
  return f.v;
}

__device__ __forceinline__ v8f wmma_bf16(v16bf a, v16bf b, v8f c) {
  return __builtin_amdgcn_wmma_f32_16x16x32_bf16(false, a, false, b, (short)0, c, false, false);
}

__device__ __forceinline__ v8f v8f_zero() {
  v8f z = {0.f, 0.f, 0.f, 0.f, 0.f, 0.f, 0.f, 0.f};
  return z;
}

// ---------------------------------------------------------------------------
// Kernel 0: A_h = Wq[h]^T @ Wk[h]   [128x128], bf16 out.  grid=H, block=256.
// ---------------------------------------------------------------------------
__global__ __launch_bounds__(256) void k0_precompute_ah(const float* __restrict__ Wq,
                                                        const float* __restrict__ Wk,
                                                        __bf16* __restrict__ Ah) {
  __shared__ __attribute__((aligned(16))) __bf16 WqT[D_DIM * 72];
  __shared__ __attribute__((aligned(16))) __bf16 WkT[D_DIM * 72];
  const int h = blockIdx.x;
  const int tid = threadIdx.x;
  const int w = tid >> 5;
  const int lane = tid & 31, hi = lane >> 4, cn = lane & 15;

  // 128-bit global reads (coalesced on d), transposed bf16 scatter to LDS.
  for (int idx = tid; idx < (KQ_DIM * D_DIM) / 4; idx += 256) {
    const int kq = idx >> 5;                 // 32 float4 per [D]-row
    const int c4 = (idx & 31) * 4;
    const v4f q = *(const v4f*)(Wq + h * KQ_DIM * D_DIM + kq * D_DIM + c4);
    const v4f k = *(const v4f*)(Wk + h * KQ_DIM * D_DIM + kq * D_DIM + c4);
#pragma unroll
    for (int j = 0; j < 4; ++j) {
      WqT[(c4 + j) * 72 + kq] = (__bf16)q[j];
      WkT[(c4 + j) * 72 + kq] = (__bf16)k[j];
    }
  }
  __syncthreads();

  v8f acc[8];
#pragma unroll
  for (int t = 0; t < 8; ++t) acc[t] = v8f_zero();

#pragma unroll
  for (int k0 = 0; k0 < KQ_DIM; k0 += 32) {
    v16bf a = load_frag_a(WqT, 72, w * 16, k0);
#pragma unroll
    for (int t = 0; t < 8; ++t) {
      v16bf bb = load_frag_b(WkT, 72, t * 16, k0);
      acc[t] = wmma_bf16(a, bb, acc[t]);
    }
  }
#pragma unroll
  for (int t = 0; t < 8; ++t) {
#pragma unroll
    for (int v = 0; v < 8; ++v) {
      const int row = w * 16 + v + 8 * hi;
      const int col = t * 16 + cn;
      Ah[h * D_DIM * D_DIM + row * D_DIM + col] = (__bf16)acc[t][v];
    }
  }
}

// ---------------------------------------------------------------------------
// Kernel 1: Gpart[s][b] = x[b, s-slab]^T @ x[b, s-slab]   fp32 partials.
// grid = B*NSPLIT, block = 256.  Double-buffered LDS + software pipeline:
// next chunk's global loads are issued before this chunk's WMMAs.
// ---------------------------------------------------------------------------
__global__ __launch_bounds__(256) void k1_gram_partial(const float* __restrict__ x,
                                                       float* __restrict__ Gpart) {
  __shared__ __attribute__((aligned(16))) __bf16 xT[2][D_DIM * 72]; // x^T tiles [d][64 n]
  const int b = blockIdx.x >> 2;
  const int s = blockIdx.x & 3;
  const int tid = threadIdx.x;
  const int w = tid >> 5;
  const int lane = tid & 31, hi = lane >> 4, cn = lane & 15;
  const float* xb = x + (size_t)(b * N_LEN + s * 512) * D_DIM;

  v8f acc[8];
#pragma unroll
  for (int t = 0; t < 8; ++t) acc[t] = v8f_zero();

  // Fixed per-thread staging slots: 8 float4 per chunk (64x128 elems / 256 thr)
  v4f reg[8];
#pragma unroll
  for (int i = 0; i < 8; ++i) {
    const int idx = tid + i * 256;
    reg[i] = *(const v4f*)(xb + (idx >> 5) * D_DIM + (idx & 31) * 4);
  }
#pragma unroll
  for (int i = 0; i < 8; ++i) {
    const int idx = tid + i * 256;
    const int nl = idx >> 5, c4 = (idx & 31) * 4;
#pragma unroll
    for (int j = 0; j < 4; ++j) xT[0][(c4 + j) * 72 + nl] = (__bf16)reg[i][j];
  }
  __syncthreads();

  for (int chunk = 0; chunk < 8; ++chunk) {
    const __bf16* cur = xT[chunk & 1];
    __bf16* nxt = xT[(chunk + 1) & 1];
    // Issue next chunk's global loads first (latency overlapped with WMMAs).
    if (chunk < 7) {
      const float* xc = xb + (size_t)(chunk + 1) * 64 * D_DIM;
#pragma unroll
      for (int i = 0; i < 8; ++i) {
        const int idx = tid + i * 256;
        reg[i] = *(const v4f*)(xc + (idx >> 5) * D_DIM + (idx & 31) * 4);
      }
    }
    // Compute current chunk.
#pragma unroll
    for (int k0 = 0; k0 < 64; k0 += 32) {
      v16bf a = load_frag_a(cur, 72, w * 16, k0);
#pragma unroll
      for (int t = 0; t < 8; ++t) {
        v16bf bb = load_frag_b(cur, 72, t * 16, k0);             // Gram: same buffer
        acc[t] = wmma_bf16(a, bb, acc[t]);
      }
    }
    // Convert + store next chunk into the alternate buffer.
    if (chunk < 7) {
#pragma unroll
      for (int i = 0; i < 8; ++i) {
        const int idx = tid + i * 256;
        const int nl = idx >> 5, c4 = (idx & 31) * 4;
#pragma unroll
        for (int j = 0; j < 4; ++j) nxt[(c4 + j) * 72 + nl] = (__bf16)reg[i][j];
      }
    }
    __syncthreads();
  }

  float* Gb = Gpart + (size_t)(s * B_NUM + b) * D_DIM * D_DIM;
#pragma unroll
  for (int t = 0; t < 8; ++t) {
#pragma unroll
    for (int v = 0; v < 8; ++v) {
      Gb[(w * 16 + v + 8 * hi) * D_DIM + t * 16 + cn] = acc[t][v];
    }
  }
}

// ---------------------------------------------------------------------------
// Kernel 2: per (h,b):  T = A_h @ G ;  Mpart[b,h] = (T @ Wv_h^T)^T   (bf16).
// G is symmetric -> natural staging works for WMMA-B.  54KB LDS pool, T
// aliased over the gemm-1 staging buffers.  grid = (H, B), block = 256.
// ---------------------------------------------------------------------------
__global__ __launch_bounds__(256) void k2_mpart(const float* __restrict__ Wv,
                                                const float* __restrict__ Gpart,
                                                const __bf16* __restrict__ Ah,
                                                __bf16* __restrict__ Mpart) {
  __shared__ __attribute__((aligned(16))) __bf16 pool[128 * 72 * 3]; // 55296 B
  __bf16* bufA = pool;          // [128][72]  A_h k-half (gemm1 A)
  __bf16* bufB = pool + 9216;   // [128][72]  G   k-half (gemm1 B)
  __bf16* bufT = pool;          // [128][136] T (17408 elems, aliases bufA+bufB)
  __bf16* bufW = pool + 18432;  // [128][72]  Wv  k-half (gemm2 B)

  const int h = blockIdx.x;
  const int b = blockIdx.y;
  const int tid = threadIdx.x;
  const int w = tid >> 5;
  const int lane = tid & 31, hi = lane >> 4, cn = lane & 15;

  // ---- gemm1: T = A_h @ G (K = 128, staged in two 64-wide halves) ----
  v8f accT[8];
#pragma unroll
  for (int t = 0; t < 8; ++t) accT[t] = v8f_zero();

  for (int kh = 0; kh < 2; ++kh) {
    // bufA: raw bf16 b128 -> b128 copy (8 elems per iteration)
    for (int idx = tid; idx < (128 * 64) / 8; idx += 256) {
      const int r = idx >> 3, c8 = (idx & 7) * 8;
      *(v4u*)(bufA + r * 72 + c8) =
          *(const v4u*)(Ah + h * 16384 + r * 128 + kh * 64 + c8);
    }
    // bufB: sum fp32 partials (float4 reads), pack 4 bf16, 8B LDS store
    for (int idx = tid; idx < (128 * 64) / 4; idx += 256) {
      const int r = idx >> 4, c4 = (idx & 15) * 4;
      v4f g = {0.f, 0.f, 0.f, 0.f};
#pragma unroll
      for (int s = 0; s < NSPLIT; ++s)
        g += *(const v4f*)(Gpart + (size_t)(s * B_NUM + b) * 16384 + r * 128 + kh * 64 + c4);
      Pack4 p;
#pragma unroll
      for (int j = 0; j < 4; ++j) p.h[j] = (__bf16)g[j];
      *(v2u*)(bufB + r * 72 + c4) = p.u;
    }
    __syncthreads();
#pragma unroll
    for (int k0 = 0; k0 < 64; k0 += 32) {
      v16bf a = load_frag_a(bufA, 72, w * 16, k0);
#pragma unroll
      for (int t = 0; t < 8; ++t) {
        v16bf bb = load_frag_b(bufB, 72, t * 16, k0);
        accT[t] = wmma_bf16(a, bb, accT[t]);
      }
    }
    __syncthreads();
  }
  // store T natural [d1][d2] (K of gemm2 contiguous)
#pragma unroll
  for (int t = 0; t < 8; ++t) {
#pragma unroll
    for (int v = 0; v < 8; ++v)
      bufT[(w * 16 + v + 8 * hi) * 136 + t * 16 + cn] = (__bf16)accT[t][v];
  }
  __syncthreads();

  // ---- gemm2: M = T @ Wv_h^T  (Wv natural [e][d] row-major == B^T staging) ----
  v8f accM[8];
#pragma unroll
  for (int t = 0; t < 8; ++t) accM[t] = v8f_zero();

  for (int kh = 0; kh < 2; ++kh) {
    for (int idx = tid; idx < (128 * 64) / 4; idx += 256) {
      const int r = idx >> 4, c4 = (idx & 15) * 4;
      const v4f f = *(const v4f*)(Wv + (size_t)(h * 128 + r) * 128 + kh * 64 + c4);
      Pack4 p;
#pragma unroll
      for (int j = 0; j < 4; ++j) p.h[j] = (__bf16)f[j];
      *(v2u*)(bufW + r * 72 + c4) = p.u;
    }
    __syncthreads();
#pragma unroll
    for (int k0 = 0; k0 < 64; k0 += 32) {
      v16bf a = load_frag_a(bufT, 136, w * 16, kh * 64 + k0);
#pragma unroll
      for (int t = 0; t < 8; ++t) {
        v16bf bb = load_frag_b(bufW, 72, t * 16, k0);
        accM[t] = wmma_bf16(a, bb, accM[t]);
      }
    }
    __syncthreads();
  }
  // write M^T partial -> [e][d1], ready for kernel-3 B staging
  __bf16* Mp = Mpart + (size_t)(b * H_NUM + h) * 16384;
#pragma unroll
  for (int t = 0; t < 8; ++t) {
#pragma unroll
    for (int v = 0; v < 8; ++v)
      Mp[(t * 16 + cn) * 128 + (w * 16 + v + 8 * hi)] = (__bf16)accM[t][v];
  }
}

// ---------------------------------------------------------------------------
// Kernel 3: out[b] = x[b] @ M[b], M summed over heads during staging.
// grid = (32 n-strips of 64, B), block = 256 (8 waves: 4 row-strips x 2 halves)
// ---------------------------------------------------------------------------
__global__ __launch_bounds__(256) void k3_out(const float* __restrict__ x,
                                              const __bf16* __restrict__ Mpart,
                                              float* __restrict__ out) {
  __shared__ __attribute__((aligned(16))) __bf16 Xl[64 * 136];   // x tile  [n][d]
  __shared__ __attribute__((aligned(16))) __bf16 Ml[128 * 136];  // M^T     [e][d]
  const int n0 = blockIdx.x * 64;
  const int b = blockIdx.y;
  const int tid = threadIdx.x;
  const int w = tid >> 5;
  const int lane = tid & 31, hi = lane >> 4, cn = lane & 15;
  const int rs = (w >> 1) * 16;     // row-strip within the 64-row tile
  const int thalf = (w & 1) * 4;    // which 4 of the 8 e-tiles

  // Ml: per 8-element chunk, one b128 load per head, fp32 sum, one b128 store.
  for (int idx = tid; idx < (D_DIM * D_DIM) / 8; idx += 256) {
    const int r = idx >> 4, c8 = (idx & 15) * 8;
    float m[8] = {0.f, 0.f, 0.f, 0.f, 0.f, 0.f, 0.f, 0.f};
#pragma unroll
    for (int h = 0; h < H_NUM; ++h) {
      Pack8 q;
      q.u = *(const v4u*)(Mpart + (size_t)(b * H_NUM + h) * 16384 + r * 128 + c8);
#pragma unroll
      for (int j = 0; j < 8; ++j) m[j] += (float)q.h[j];
    }
    Pack8 p;
#pragma unroll
    for (int j = 0; j < 8; ++j) p.h[j] = (__bf16)m[j];
    *(v4u*)(Ml + r * 136 + c8) = p.u;
  }
  // Xl: float4 coalesced reads, packed 8B LDS stores (natural layout).
  for (int idx = tid; idx < (64 * D_DIM) / 4; idx += 256) {
    const int r = idx >> 5, c4 = (idx & 31) * 4;
    const v4f f = *(const v4f*)(x + (size_t)(b * N_LEN + n0 + r) * D_DIM + c4);
    Pack4 p;
#pragma unroll
    for (int j = 0; j < 4; ++j) p.h[j] = (__bf16)f[j];
    *(v2u*)(Xl + r * 136 + c4) = p.u;
  }
  __syncthreads();

  v8f acc[4];
#pragma unroll
  for (int t = 0; t < 4; ++t) acc[t] = v8f_zero();

#pragma unroll
  for (int k0 = 0; k0 < D_DIM; k0 += 32) {
    v16bf a = load_frag_a(Xl, 136, rs, k0);
#pragma unroll
    for (int tt = 0; tt < 4; ++tt) {
      v16bf bb = load_frag_b(Ml, 136, (thalf + tt) * 16, k0);
      acc[tt] = wmma_bf16(a, bb, acc[tt]);
    }
  }
#pragma unroll
  for (int tt = 0; tt < 4; ++tt) {
#pragma unroll
    for (int v = 0; v < 8; ++v) {
      const int n = n0 + rs + v + 8 * hi;
      const int e = (thalf + tt) * 16 + cn;
      out[(size_t)(b * N_LEN + n) * D_DIM + e] = acc[tt][v];
    }
  }
}

// ---------------------------------------------------------------------------
extern "C" void kernel_launch(void* const* d_in, const int* in_sizes, int n_in,
                              void* d_out, int out_size, void* d_ws, size_t ws_size,
                              hipStream_t stream) {
  (void)in_sizes; (void)n_in; (void)out_size; (void)ws_size;
  const float* x  = (const float*)d_in[0];
  const float* Wk = (const float*)d_in[1];
  const float* Wq = (const float*)d_in[2];
  const float* Wv = (const float*)d_in[3];
  float* out = (float*)d_out;

  char* ws = (char*)d_ws;
  // ws layout: Gpart fp32 [4][32][128*128] = 8 MB | Ah bf16 [8][128*128] = 256 KB
  //            | Mpart bf16 [32*8][128*128] = 8 MB   (total ~16.25 MB)
  float*  Gpart = (float*)ws;
  __bf16* Ah    = (__bf16*)(ws + (size_t)8 * 1024 * 1024);
  __bf16* Mpart = (__bf16*)(ws + (size_t)8 * 1024 * 1024 + (size_t)256 * 1024);

  k0_precompute_ah<<<H_NUM, 256, 0, stream>>>(Wq, Wk, Ah);
  k1_gram_partial<<<B_NUM * NSPLIT, 256, 0, stream>>>(x, Gpart);
  k2_mpart<<<dim3(H_NUM, B_NUM), 256, 0, stream>>>(Wv, Gpart, Ah, Mpart);
  k3_out<<<dim3(32, B_NUM), 256, 0, stream>>>(x, Mpart, out);
}
